// ProductQuantizerSimple_46660524704250
// MI455X (gfx1250) — matrix operand order repaired and checked by reading, not compiled, not run on hardware
//
#include <hip/hip_runtime.h>
#include <hip/hip_bf16.h>

#define H_NUM 8
#define CB_K 1024
#define D_DIM 128

typedef __attribute__((ext_vector_type(16))) __bf16 v16bf;
typedef __attribute__((ext_vector_type(8)))  float  v8f;
typedef __attribute__((ext_vector_type(4)))  unsigned int u32x4;
typedef __attribute__((ext_vector_type(4)))  float  f32x4;

union V16 { v16bf v; u32x4 q[2]; };

// ---------------- prep: codebook -> bf16 hi/lo split + c_sq ----------------
__global__ __launch_bounds__(D_DIM) void pq_prep(const float* __restrict__ cb,
                                                 __bf16* __restrict__ cb_hi,
                                                 __bf16* __restrict__ cb_lo,
                                                 float* __restrict__ c_sq) {
  const int code = blockIdx.x;      // 0 .. H*K-1
  const int d    = threadIdx.x;     // 0 .. 127
  float x = cb[(size_t)code * D_DIM + d];
  __bf16 hi = (__bf16)x;
  __bf16 lo = (__bf16)(x - (float)hi);
  cb_hi[(size_t)code * D_DIM + d] = hi;
  cb_lo[(size_t)code * D_DIM + d] = lo;
  float s = x * x;
  #pragma unroll
  for (int off = 16; off >= 1; off >>= 1) s += __shfl_xor(s, off, 32);
  __shared__ float part[4];
  if ((threadIdx.x & 31) == 0) part[threadIdx.x >> 5] = s;
  __syncthreads();
  if (threadIdx.x == 0) c_sq[code] = part[0] + part[1] + part[2] + part[3];
}

// ---------------- main: WMMA cross-term + argmin + gather ----------------
// 128 threads = 4 waves, 64 batch rows per block. Wave g owns code tiles
// t = g, g+4, ... Each wave computes FOUR 16-row batch subtiles per B load
// (B register reuse -> 1/4 the L2 traffic) with 2 accumulator chains per
// subtile (hi*hi; hi*lo + lo*hi) -> 8 independent WMMA chains, 48 WMMAs per
// 8KB of B-operand loads.
__global__ __launch_bounds__(128) void pq_main(const float* __restrict__ z,
                                               const float* __restrict__ cb,
                                               const __bf16* __restrict__ cb_hi,
                                               const __bf16* __restrict__ cb_lo,
                                               const float* __restrict__ c_sq,
                                               float* __restrict__ zq,
                                               int* __restrict__ idx_out) {
  constexpr int STR  = D_DIM + 8;                // LDS row stride (bf16) -> bank spread
  constexpr int ROWS = 64;                       // batch rows per block
  __shared__ __align__(16) __bf16 lds_hi[ROWS * STR];
  __shared__ __align__(16) __bf16 lds_lo[ROWS * STR];
  __shared__ float redV[ROWS][4];
  __shared__ int   redI[ROWS][4];
  __shared__ int   fidx[ROWS];

  const int h   = blockIdx.y;
  const int b0  = blockIdx.x * ROWS;
  const int tid = threadIdx.x;

  // ---- load z tile (64 rows x 128 f32), split to bf16 hi/lo in LDS ----
  #pragma unroll
  for (int j = 0; j < 16; ++j) {
    int i   = tid + 128 * j;                     // float4 index 0..2047
    int row = i >> 5;
    int d4  = i & 31;
    f32x4 v = *(const f32x4*)(z + (size_t)(b0 + row) * (H_NUM * D_DIM) + h * D_DIM + d4 * 4);
    #pragma unroll
    for (int e = 0; e < 4; ++e) {
      float x = v[e];
      __bf16 hi = (__bf16)x;
      __bf16 lo = (__bf16)(x - (float)hi);
      lds_hi[row * STR + d4 * 4 + e] = hi;
      lds_lo[row * STR + d4 * 4 + e] = lo;
    }
  }
  __syncthreads();

  const int g    = tid >> 5;        // wave id = code-tile group 0..3
  const int lane = tid & 31;
  const int half = lane >> 4;       // lane half
  const int m    = lane & 15;

  // ---- hoist A operands for all 4 subtiles (contiguous K order: K = 16*half + element, same as B) ----
  v16bf aHi[4][4], aLo[4][4];
  #pragma unroll
  for (int sub = 0; sub < 4; ++sub) {
    const int arow = (sub * 16 + m) * STR;
    #pragma unroll
    for (int kc = 0; kc < 4; ++kc) {
      V16 th, tl;
      int off0 = arow + kc * 32 + half * 16;     // K = kc*32 + half*16 .. +15
      th.q[0] = *(const u32x4*)(lds_hi + off0);
      th.q[1] = *(const u32x4*)(lds_hi + off0 + 8);   // K = +8 chunk
      tl.q[0] = *(const u32x4*)(lds_lo + off0);
      tl.q[1] = *(const u32x4*)(lds_lo + off0 + 8);
      aHi[sub][kc] = th.v; aLo[sub][kc] = tl.v;
    }
  }

  float bestVal[4][8];
  int   bestIdx[4][8];
  #pragma unroll
  for (int s = 0; s < 4; ++s)
    #pragma unroll
    for (int r = 0; r < 8; ++r) { bestVal[s][r] = 3.4e38f; bestIdx[s][r] = 0; }

  // ---- sweep 16 code tiles per wave, 48 WMMAs per tile ----
  for (int t = g; t < CB_K / 16; t += 4) {
    const int code = t * 16 + m;                 // lane's code column (N = lane%16)
    const __bf16* bh = cb_hi + (((size_t)h * CB_K + code) * D_DIM + half * 16);
    const __bf16* bl = cb_lo + (((size_t)h * CB_K + code) * D_DIM + half * 16);
    const v8f zero = {0.f, 0.f, 0.f, 0.f, 0.f, 0.f, 0.f, 0.f};
    v8f accA[4] = {zero, zero, zero, zero};      // hi*hi chains
    v8f accB[4] = {zero, zero, zero, zero};      // hi*lo + lo*hi chains
    #pragma unroll
    for (int kc = 0; kc < 4; ++kc) {
      V16 vbh, vbl;
      vbh.q[0] = *(const u32x4*)(bh + kc * 32);       // B: 16 contiguous K per lane
      vbh.q[1] = *(const u32x4*)(bh + kc * 32 + 8);
      vbl.q[0] = *(const u32x4*)(bl + kc * 32);
      vbl.q[1] = *(const u32x4*)(bl + kc * 32 + 8);
      #pragma unroll
      for (int sub = 0; sub < 4; ++sub)
        accA[sub] = __builtin_amdgcn_wmma_f32_16x16x32_bf16(false, aHi[sub][kc], false, vbh.v,
                                                            (short)0, accA[sub], false, false);
      #pragma unroll
      for (int sub = 0; sub < 4; ++sub)
        accB[sub] = __builtin_amdgcn_wmma_f32_16x16x32_bf16(false, aHi[sub][kc], false, vbl.v,
                                                            (short)0, accB[sub], false, false);
      #pragma unroll
      for (int sub = 0; sub < 4; ++sub)
        accB[sub] = __builtin_amdgcn_wmma_f32_16x16x32_bf16(false, aLo[sub][kc], false, vbh.v,
                                                            (short)0, accB[sub], false, false);
    }
    float cs = c_sq[h * CB_K + code];
    #pragma unroll
    for (int sub = 0; sub < 4; ++sub) {
      #pragma unroll
      for (int r = 0; r < 8; ++r) {
        float s  = accA[sub][r] + accB[sub][r];
        float d2 = __builtin_fmaf(-2.0f, s, cs);     // z_sq constant -> dropped
        if (d2 < bestVal[sub][r]) { bestVal[sub][r] = d2; bestIdx[sub][r] = code; }
      }
    }
  }

  // ---- reduce argmin across the 16 code-lanes (wave32 butterfly) ----
  #pragma unroll
  for (int sub = 0; sub < 4; ++sub) {
    #pragma unroll
    for (int r = 0; r < 8; ++r) {
      float v = bestVal[sub][r]; int ii = bestIdx[sub][r];
      #pragma unroll
      for (int off = 1; off < 16; off <<= 1) {
        float ov = __shfl_xor(v, off, 32);
        int   oi = __shfl_xor(ii, off, 32);
        if (ov < v || (ov == v && oi < ii)) { v = ov; ii = oi; }
      }
      if (m == 0) {
        int row = sub * 16 + half * 8 + r;       // C/D layout: VGPR r -> row r + 8*half
        redV[row][g] = v; redI[row][g] = ii;
      }
    }
  }
  __syncthreads();

  // ---- reduce across the 4 code groups, emit idx ----
  if (tid < ROWS) {
    float v = redV[tid][0]; int ii = redI[tid][0];
    #pragma unroll
    for (int gg = 1; gg < 4; ++gg) {
      float ov = redV[tid][gg]; int oi = redI[tid][gg];
      if (ov < v || (ov == v && oi < ii)) { v = ov; ii = oi; }
    }
    fidx[tid] = ii;
    idx_out[(size_t)(b0 + tid) * H_NUM + h] = ii;
  }
  __syncthreads();

  // ---- exact f32 gather of winning codebook rows -> zq ----
  #pragma unroll
  for (int j = 0; j < 16; ++j) {
    int i   = tid + 128 * j;
    int row = i >> 5;
    int d4  = i & 31;
    int ci  = fidx[row];
    f32x4 v = *(const f32x4*)(cb + ((size_t)h * CB_K + ci) * D_DIM + d4 * 4);
    *(f32x4*)(zq + (size_t)(b0 + row) * (H_NUM * D_DIM) + h * D_DIM + d4 * 4) = v;
  }
}

extern "C" void kernel_launch(void* const* d_in, const int* in_sizes, int n_in,
                              void* d_out, int out_size, void* d_ws, size_t ws_size,
                              hipStream_t stream) {
  const float* z  = (const float*)d_in[0];
  const float* cb = (const float*)d_in[1];
  const int B = in_sizes[0] / (H_NUM * D_DIM);

  __bf16* cb_hi = (__bf16*)d_ws;
  __bf16* cb_lo = cb_hi + (size_t)H_NUM * CB_K * D_DIM;
  float*  c_sq  = (float*)(cb_lo + (size_t)H_NUM * CB_K * D_DIM);

  float* zq      = (float*)d_out;
  int*   idx_out = (int*)(zq + (size_t)B * H_NUM * D_DIM);

  pq_prep<<<H_NUM * CB_K, D_DIM, 0, stream>>>(cb, cb_hi, cb_lo, c_sq);

  dim3 grid(B / 64, H_NUM);
  pq_main<<<grid, 128, 0, stream>>>(z, cb, cb_hi, cb_lo, c_sq, zq, idx_out);
}
